// DeformConv_2d_29927332119113
// MI455X (gfx1250) — matrix operand order repaired and checked by reading, not compiled, not run on hardware
//
#include <hip/hip_runtime.h>

typedef __attribute__((ext_vector_type(16))) __bf16 v16bf;
typedef __attribute__((ext_vector_type(8)))  float  v8f;
typedef __attribute__((ext_vector_type(2)))  float  v2f;

#define B_    8
#define CIN   64
#define H_    96
#define W_    96
#define O_    128
#define KK    9
#define HW    (H_ * W_)          // 9216
#define KDIM  (CIN * KK)         // 576
#define NTOT  (B_ * HW)          // 73728
#define NTILE 32

// ---------------------------------------------------------------------------
// Kernel 0: NCHW -> NHWC transpose of x (f32), LDS-tiled for coalescing.
// ---------------------------------------------------------------------------
__global__ void __launch_bounds__(256)
k_transpose_x(const float* __restrict__ x, float* __restrict__ xn) {
    __shared__ float t[CIN][33];
    int blk = blockIdx.x;                 // 8*96*3 = 2304 blocks
    int wsl = blk % 3;
    int ho  = (blk / 3) % H_;
    int b   = blk / (3 * H_);
    int wo0 = wsl * 32;
    int tid = threadIdx.x;

    int lw = tid & 31, cg = tid >> 5;     // load: 32 wo-lanes x 8 c-groups
#pragma unroll
    for (int r = 0; r < 8; ++r) {
        int c = cg + r * 8;
        t[c][lw] = x[((size_t)(b * CIN + c) * H_ + ho) * W_ + wo0 + lw];
    }
    __syncthreads();
    int c2 = tid & 63, pg = tid >> 6;     // store: 64 c-lanes x 4 pixel-groups
#pragma unroll
    for (int r = 0; r < 8; ++r) {
        int wl = pg + r * 4;
        xn[(size_t)(b * HW + ho * W_ + wo0 + wl) * CIN + c2] = t[c2][wl];
    }
}

// ---------------------------------------------------------------------------
// Kernel 1: weight preparation.
//   wgt  (bf16): [o][kk*64 + c]   <- weight[o][c][kk]  (main GEMM A, K-major)
//   wgt2 (f32) : [ch][kk*64 + c]  <- concat(offset_w, mask_w), ch 27..31 = 0
// ---------------------------------------------------------------------------
__global__ void __launch_bounds__(256)
k_prep_w(const float* __restrict__ offw, const float* __restrict__ maskw,
         const float* __restrict__ weight,
         __bf16* __restrict__ wgt, float* __restrict__ wgt2) {
    int idx = blockIdx.x * 256 + threadIdx.x;
    if (idx < O_ * KDIM) {
        int o = idx / KDIM, kc = idx - o * KDIM;
        int kk = kc >> 6, c = kc & 63;
        wgt[idx] = (__bf16)weight[(o * CIN + c) * KK + kk];
    }
    if (idx < 32 * KDIM) {
        int ch = idx / KDIM, kc = idx - ch * KDIM;
        int kk = kc >> 6, c = kc & 63;
        float v = 0.0f;
        if (ch < 18)      v = offw[(ch * CIN + c) * KK + kk];
        else if (ch < 27) v = maskw[((ch - 18) * CIN + c) * KK + kk];
        wgt2[idx] = v;
    }
}

// ---------------------------------------------------------------------------
// Kernel 2: offset+mask 3x3 conv via V_WMMA_F32_16X16X4_F32 (full f32).
// Implicit GEMM: M=32 (27 channels padded), N=NTOT, K=576.
// Block tile: M=32 x N=64; 8 waves -> 2x4 C-tiles of 16x16; no LDS.
// A/B fragments: 2 f32/lane loaded straight from L2-resident global.
// ---------------------------------------------------------------------------
__global__ void __launch_bounds__(256)
k_offmask_wmma(const float* __restrict__ xn, const float* __restrict__ wgt2,
               const float* __restrict__ offb, const float* __restrict__ maskb,
               float* __restrict__ om) {
    const int tid  = threadIdx.x;
    const int wave = tid >> 5, lane = tid & 31;
    const int l16  = lane & 15, hi = lane >> 4;
    const int mt   = wave & 1;            // m-tile 0..1
    const int nt   = wave >> 1;           // n-tile 0..3

    const int ns  = blockIdx.x * 64 + nt * 16 + l16;
    const int b   = ns / HW;
    const int rem = ns - b * HW;
    const int ho  = rem / W_;
    const int wo  = rem - ho * W_;
    const float* xb = xn + (size_t)b * HW * CIN;

    const int mb = mt * 16;
    // A (16x4 f32): lanes 0-15 hold K{k0,k0+1} of row M=lane; lanes 16-31 K{k0+2,k0+3}
    const float* arow = wgt2 + (size_t)(mb + l16) * KDIM + hi * 2;

    v8f acc = {0.f,0.f,0.f,0.f,0.f,0.f,0.f,0.f};

    for (int kk = 0; kk < KK; ++kk) {
        int ky = kk / 3, kx = kk - ky * 3;
        int y = ho - 1 + ky, x = wo - 1 + kx;
        float vmask = (y >= 0 && y < H_ && x >= 0 && x < W_) ? 1.0f : 0.0f;
        int yc = min(max(y, 0), H_ - 1), xc = min(max(x, 0), W_ - 1);
        // B (4x16 f32): lanes 0-15 hold K{k0,k0+1} of col N=lane; lanes 16-31 K{k0+2,k0+3}
        const float* bpix = xb + (size_t)(yc * W_ + xc) * CIN + hi * 2;
        const float* apix = arow + kk * 64;
#pragma unroll
        for (int c0 = 0; c0 < 64; c0 += 4) {
            v2f av = *(const v2f*)(apix + c0);
            v2f bv = *(const v2f*)(bpix + c0);
            bv.x *= vmask;                 // zero-padding, branch-free (EXEC stays full)
            bv.y *= vmask;
            acc = __builtin_amdgcn_wmma_f32_16x16x4_f32(
                false, av, false, bv, (short)0, acc, false, false);
        }
    }

    // C layout: VGPR r -> M=r (lanes<16) / M=r+8 (lanes>=16), N=lane%16
    float* ob = om + (size_t)b * 27 * HW + rem;
#pragma unroll
    for (int r = 0; r < 8; ++r) {
        int ch = mb + r + 8 * hi;
        if (ch < 27) {
            float bia = (ch < 18) ? offb[ch] : maskb[ch - 18];
            ob[(size_t)ch * HW] = acc[r] + bia;
        }
    }
}

// ---------------------------------------------------------------------------
// Kernel 3: fused deformable gather + bf16 WMMA implicit GEMM.
// Block tile: M = 128 (all O), N = 32 spatial positions. K = 576.
// ---------------------------------------------------------------------------
struct Samp { int o00, o01, o10, o11; float w00, w01, w10, w11; };

__global__ void __launch_bounds__(256)
k_deform_wmma(const float* __restrict__ xn, const float* __restrict__ om,
              const __bf16* __restrict__ wgt, const float* __restrict__ bias,
              float* __restrict__ out) {
    __shared__ Samp samp[NTILE * KK];                       // 9216 B
    __shared__ __align__(16) __bf16 cols[NTILE][KDIM];      // 36864 B

    const int tid   = threadIdx.x;
    const int nbase = blockIdx.x * NTILE;

    // ---- Phase 1: per-(position, tap) sample descriptors -------------------
    for (int e = tid; e < NTILE * KK; e += 256) {
        int n   = e / KK, tap = e - n * KK;
        int ns  = nbase + n;
        int b   = ns / HW;
        int rem = ns - b * HW;
        int ho  = rem / W_;
        int wo  = rem - ho * W_;
        int ky  = tap / 3, kx = tap - ky * 3;

        const float* omb = om + (size_t)b * 27 * HW + rem;
        float dy = omb[(size_t)(2 * tap)     * HW];
        float dx = omb[(size_t)(2 * tap + 1) * HW];
        float mr = omb[(size_t)(18 + tap)    * HW];
        float m  = 1.0f / (1.0f + __expf(-mr));             // sigmoid(mask)

        float py = (float)(ho - 1 + ky) + dy;
        float px = (float)(wo - 1 + kx) + dx;
        float y0f = floorf(py), x0f = floorf(px);
        float fy = py - y0f, fx = px - x0f;
        int y0 = (int)y0f, x0 = (int)x0f;
        int y1 = y0 + 1,   x1 = x0 + 1;
        bool vy0 = (y0 >= 0) && (y0 < H_), vy1 = (y1 >= 0) && (y1 < H_);
        bool vx0 = (x0 >= 0) && (x0 < W_), vx1 = (x1 >= 0) && (x1 < W_);

        Samp s;
        s.w00 = (vy0 && vx0) ? (1.0f - fy) * (1.0f - fx) * m : 0.0f;
        s.w01 = (vy0 && vx1) ? (1.0f - fy) * fx * m          : 0.0f;
        s.w10 = (vy1 && vx0) ? fy * (1.0f - fx) * m          : 0.0f;
        s.w11 = (vy1 && vx1) ? fy * fx * m                   : 0.0f;
        int y0c = min(max(y0, 0), H_ - 1), y1c = min(max(y1, 0), H_ - 1);
        int x0c = min(max(x0, 0), W_ - 1), x1c = min(max(x1, 0), W_ - 1);
        s.o00 = y0c * W_ + x0c;  s.o01 = y0c * W_ + x1c;
        s.o10 = y1c * W_ + x0c;  s.o11 = y1c * W_ + x1c;
        samp[e] = s;
    }
    __syncthreads();

    // ---- Phase 2: bilinear gather -> bf16 cols tile, 4 channels/thread -----
    for (int g = tid; g < NTILE * (KDIM / 4); g += 256) {   // 18 iterations
        int n  = g / (KDIM / 4);
        int r  = g - n * (KDIM / 4);
        int kk = r >> 4;
        int c4 = (r & 15) * 4;
        int ns = nbase + n;
        int b  = ns / HW;
        const float* xb = xn + (size_t)b * HW * CIN + c4;
        Samp s = samp[n * KK + kk];
        float4 p00 = *(const float4*)(xb + (size_t)s.o00 * CIN);
        float4 p01 = *(const float4*)(xb + (size_t)s.o01 * CIN);
        float4 p10 = *(const float4*)(xb + (size_t)s.o10 * CIN);
        float4 p11 = *(const float4*)(xb + (size_t)s.o11 * CIN);
        union { __bf16 h[4]; uint2 u; } pk;
        pk.h[0] = (__bf16)(s.w00*p00.x + s.w01*p01.x + s.w10*p10.x + s.w11*p11.x);
        pk.h[1] = (__bf16)(s.w00*p00.y + s.w01*p01.y + s.w10*p10.y + s.w11*p11.y);
        pk.h[2] = (__bf16)(s.w00*p00.z + s.w01*p01.z + s.w10*p10.z + s.w11*p11.z);
        pk.h[3] = (__bf16)(s.w00*p00.w + s.w01*p01.w + s.w10*p10.w + s.w11*p11.w);
        *(uint2*)(&cols[n][kk * 64 + c4]) = pk.u;
    }
    __syncthreads();

    // ---- Phase 3: WMMA K-loop ---------------------------------------------
    const int wave = tid >> 5, lane = tid & 31;
    const int l16 = lane & 15, hi = lane >> 4;
    const int mbase = wave * 16;                  // each wave: 16 O-rows x 32 N
    v8f acc0 = {0.f,0.f,0.f,0.f,0.f,0.f,0.f,0.f};
    v8f acc1 = acc0;

    // A fragment (16x32 bf16): lanes 0-15 hold K[0..7],K[16..23] of row M=lane;
    // lanes 16-31 hold K[8..15],K[24..31] of row M=lane-16.
    const __bf16* arow = wgt + (size_t)(mbase + l16) * KDIM + hi * 8;
    // B fragment (32x16 bf16): lanes 0-15 hold K[0..7],K[16..23] of col N=lane;
    // lanes 16-31 hold K[8..15],K[24..31] of col N=lane-16.
    const __bf16* b0 = &cols[l16][hi * 8];
    const __bf16* b1 = &cols[16 + l16][hi * 8];

#pragma unroll 2
    for (int k0 = 0; k0 < KDIM; k0 += 32) {
        union { int4 i[2]; v16bf v; } fa, fb0, fb1;
        fa.i[0]  = *(const int4*)(arow + k0);        // K[k0+0..7] / K[k0+8..15]
        fa.i[1]  = *(const int4*)(arow + k0 + 16);   // K[k0+16..23] / K[k0+24..31]
        fb0.i[0] = *(const int4*)(b0 + k0);
        fb0.i[1] = *(const int4*)(b0 + k0 + 16);
        fb1.i[0] = *(const int4*)(b1 + k0);
        fb1.i[1] = *(const int4*)(b1 + k0 + 16);
        acc0 = __builtin_amdgcn_wmma_f32_16x16x32_bf16(
            false, fa.v, false, fb0.v, (short)0, acc0, false, false);
        acc1 = __builtin_amdgcn_wmma_f32_16x16x32_bf16(
            false, fa.v, false, fb1.v, (short)0, acc1, false, false);
    }

    // ---- Phase 4: epilogue -------------------------------------------------
#pragma unroll
    for (int t = 0; t < 2; ++t) {
        int ns  = nbase + t * 16 + l16;
        int b   = ns / HW;
        int rem = ns - b * HW;
        float* orow = out + (size_t)b * O_ * HW + rem;
        v8f a = t ? acc1 : acc0;
#pragma unroll
        for (int r = 0; r < 8; ++r) {
            int o = mbase + r + 8 * hi;
            orow[(size_t)o * HW] = a[r] + bias[o];
        }
    }
}

// ---------------------------------------------------------------------------
// Host launcher
// ---------------------------------------------------------------------------
extern "C" void kernel_launch(void* const* d_in, const int* in_sizes, int n_in,
                              void* d_out, int out_size, void* d_ws, size_t ws_size,
                              hipStream_t stream) {
    const float* x      = (const float*)d_in[0];
    const float* offw   = (const float*)d_in[1];
    const float* offb   = (const float*)d_in[2];
    const float* maskw  = (const float*)d_in[3];
    const float* maskb  = (const float*)d_in[4];
    const float* weight = (const float*)d_in[5];
    const float* bias   = (const float*)d_in[6];
    float* out = (float*)d_out;

    char* ws = (char*)d_ws;
    float*  xn   = (float*) (ws);               // NHWC x:      18,874,368 B
    float*  om   = (float*) (ws + 18874368);    // offset+mask:  7,962,624 B
    __bf16* wgt  = (__bf16*)(ws + 26836992);    // bf16 weight:    147,456 B
    float*  wgt2 = (float*) (ws + 26984448);    // conv weights:    73,728 B

    k_transpose_x  <<<2304, 256, 0, stream>>>(x, xn);
    k_prep_w       <<<288,  256, 0, stream>>>(offw, maskw, weight, wgt, wgt2);
    k_offmask_wmma <<<1152, 256, 0, stream>>>(xn, wgt2, offb, maskb, om);
    k_deform_wmma  <<<2304, 256, 0, stream>>>(xn, om, wgt, bias, out);
}
